// ImprovedMultiCrossAttention_7722351198477
// MI455X (gfx1250) — matrix operand order repaired and checked by reading, not compiled, not run on hardware
//
#include <hip/hip_runtime.h>
#include <cstdint>
#include <cstddef>

#define Bz 2
#define Sq 2048
#define Dm 1024
#define Hh 16
#define DhD 64
#define BSROWS (Bz * Sq)   /* 4096 */
#define MQ (2 * BSROWS)    /* 8192 */

typedef __attribute__((ext_vector_type(16))) __bf16 v16bf;
typedef __attribute__((ext_vector_type(8)))  float  v8f;
typedef __attribute__((ext_vector_type(4)))  unsigned v4u;
typedef __attribute__((ext_vector_type(8)))  int v8i;
typedef __attribute__((ext_vector_type(4)))  int v4i;

union AFrag { v16bf v; unsigned u[8]; };

__device__ inline unsigned short f2bf(float f) {
  union { float f; unsigned u; } x; x.f = f;
  unsigned r = x.u + 0x7FFFu + ((x.u >> 16) & 1u);
  return (unsigned short)(r >> 16);
}

// Pack two f32 -> two bf16 (round-half-up) in 3 VALU ops via v_perm_b32.
__device__ inline unsigned f2bf_pk(float a, float b) {
  union { float f; unsigned u; } x, y; x.f = a; y.f = b;
  return __builtin_amdgcn_perm(y.u + 0x8000u, x.u + 0x8000u, 0x07060302u);
}

__device__ inline v8f wmma_bf16(v16bf a, v16bf b, v8f c) {
  return __builtin_amdgcn_wmma_f32_16x16x32_bf16(
      false, a, false, b, (short)0, c, false, false);
}

// ---------------------------------------------------------------------------
// TDM: async 2D bf16 tile load -> LDS. D# per CDNA5 ISA ch.8.
// tile_d0 = contiguous elements per row, tile_d1 = rows,
// tens_d0/tens_d1 = tensor extents, stride0 = row stride (elements).
// ---------------------------------------------------------------------------
__device__ inline void tdm_load_2d_bf16(unsigned lds_off, unsigned long long ga,
                                        unsigned tile_d0, unsigned tile_d1,
                                        unsigned tens_d0, unsigned tens_d1,
                                        unsigned stride0) {
  v4u g0;
  g0[0] = 1u;                                           // count=1, no gather
  g0[1] = lds_off;                                      // lds_addr (bytes)
  g0[2] = (unsigned)(ga & 0xFFFFFFFFull);               // global_addr lo
  g0[3] = (unsigned)((ga >> 32) & 0x1FFFFFFull) | (2u << 30);  // hi | type=2
  v8i g1;
  g1[0] = 0x00010000;                                   // data_size=1 (2B)
  g1[1] = (int)(tens_d0 << 16);                         // tensor_dim0 lo16
  g1[2] = (int)((tens_d0 >> 16) | (tens_d1 << 16));     // td0 hi | td1 lo
  g1[3] = (int)((tens_d1 >> 16) | (tile_d0 << 16));     // td1 hi | tile_dim0
  g1[4] = (int)(tile_d1 & 0xFFFFu);                     // tile_dim1 | tile_dim2=0
  g1[5] = (int)stride0;                                 // tensor_dim0_stride lo
  g1[6] = 0;
  g1[7] = 0;
  v4i z4 = {0, 0, 0, 0};
#if __clang_major__ >= 23
  v8i z8 = {0, 0, 0, 0, 0, 0, 0, 0};
  __builtin_amdgcn_tensor_load_to_lds(g0, g1, z4, z4, z8, 0);
#else
  __builtin_amdgcn_tensor_load_to_lds(g0, g1, z4, z4, 0);
#endif
}

// ---------------------------------------------------------------------------
// Kernel 1: QKV projection. C[M=8192, N=1024] = concat(s0,s1) @ W, bf16 out.
// Double-buffered LDS, one barrier per 32-K step.
// ---------------------------------------------------------------------------
__global__ void __launch_bounds__(128)
gemm_qkv(const float* __restrict__ s0, const float* __restrict__ s1,
         const float* __restrict__ W, unsigned short* __restrict__ out)
{
  __shared__ __align__(32) unsigned short As[2][64][32];
  __shared__ __align__(32) unsigned short Bs[2][32][64];
  const int tid = threadIdx.x, lane = tid & 31, wave = tid >> 5;
  const int half = lane >> 4, lm = lane & 15;
  const int mBase = blockIdx.x * 64, nBase = blockIdx.y * 64;

  v8f acc[4] = {};

  const int arow = tid >> 1, acb = (tid & 1) << 4;   // 64 rows x 32 cols
  const int gr = mBase + arow;
  const float* asrc = (gr < BSROWS) ? (s0 + (size_t)gr * Dm)
                                    : (s1 + (size_t)(gr - BSROWS) * Dm);
  const int brow = tid >> 2, bcb = (tid & 3) << 4;   // 32 rows x 64 cols

  auto loadT = [&](int buf, int kk) {
    const float* ap = asrc + kk + acb;
    __builtin_prefetch(ap + 32, 0, 3);
    #pragma unroll
    for (int i = 0; i < 8; ++i)
      *(unsigned*)&As[buf][arow][acb + 2 * i] = f2bf_pk(ap[2 * i], ap[2 * i + 1]);
    const float* bp = W + (size_t)(kk + brow) * Dm + nBase + bcb;
    __builtin_prefetch(bp + 32 * Dm, 0, 3);
    #pragma unroll
    for (int i = 0; i < 8; ++i)
      *(unsigned*)&Bs[buf][brow][bcb + 2 * i] = f2bf_pk(bp[2 * i], bp[2 * i + 1]);
  };

  loadT(0, 0);
  __syncthreads();

  for (int kk = 0; kk < Dm; kk += 32) {
    const int cur = (kk >> 5) & 1;
    if (kk + 32 < Dm) loadT(cur ^ 1, kk + 32);

    AFrag a;
    const int m = wave * 16 + lm;
    #pragma unroll
    for (int v = 0; v < 8; ++v) {
      const int kb = ((v >> 2) << 4) + ((v & 3) << 1) + (half << 3);
      a.u[v] = *(const unsigned*)&As[cur][m][kb];
    }
    #pragma unroll
    for (int nt = 0; nt < 4; ++nt) {
      AFrag bb;
      #pragma unroll
      for (int v = 0; v < 8; ++v) {
        const int kb = ((v >> 2) << 4) + ((v & 3) << 1) + (half << 3);
        bb.u[v] = (unsigned)Bs[cur][kb][(nt << 4) + lm] | ((unsigned)Bs[cur][kb + 1][(nt << 4) + lm] << 16);
      }
      acc[nt] = wmma_bf16(a.v, bb.v, acc[nt]);
    }
    __syncthreads();
  }

  #pragma unroll
  for (int nt = 0; nt < 4; ++nt) {
    #pragma unroll
    for (int v = 0; v < 8; ++v) {
      const int r = mBase + wave * 16 + v + (half << 3);
      const int c = nBase + (nt << 4) + lm;
      out[(size_t)r * Dm + c] = f2bf(acc[nt][v]);
    }
  }
}

// ---------------------------------------------------------------------------
// Kernel 2: dual-stream flash attention. Q/K/V bf16 in ws.
// Rows [0,4096) = x-stream quantities, [4096,8192) = y-stream.
// V tiles via pipelined TDM (double-buffered LDS, tensorcnt sync);
// K tiles staged manually (need d<->key transpose).
// grid = (S/64, B*H); block = 128 (4 waves, 16 queries each).
// ---------------------------------------------------------------------------
__global__ void __launch_bounds__(128)
attn(const unsigned short* __restrict__ Q, const unsigned short* __restrict__ K,
     const unsigned short* __restrict__ V,
     unsigned short* __restrict__ A1, unsigned short* __restrict__ A2)
{
  __shared__ __align__(32) unsigned short Kxt[64][32];     // transposed [d][key]
  __shared__ __align__(32) unsigned short Kyt[64][32];
  __shared__ __align__(32) unsigned short Vxs[2][32][64];  // [buf][key][d]
  __shared__ __align__(32) unsigned short Vys[2][32][64];
  __shared__ __align__(32) unsigned short Ps[4][16][32];   // per-wave P tile

  const int tid = threadIdx.x, lane = tid & 31, wave = tid >> 5;
  const int half = lane >> 4, lm = lane & 15;
  const int b = blockIdx.y >> 4, h = blockIdx.y & 15;
  const int qBase = blockIdx.x * 64 + wave * 16;

  // LDS aperture maps addr[31:0] directly to the LDS offset (ISA 10.2).
  const unsigned ldsVx[2] = { (unsigned)(uintptr_t)(void*)&Vxs[0][0][0],
                              (unsigned)(uintptr_t)(void*)&Vxs[1][0][0] };
  const unsigned ldsVy[2] = { (unsigned)(uintptr_t)(void*)&Vys[0][0][0],
                              (unsigned)(uintptr_t)(void*)&Vys[1][0][0] };
  auto gaV = [&](int kcc, int strm) -> unsigned long long {
    return (unsigned long long)(uintptr_t)
           (V + ((size_t)(b * Sq + kcc) + (size_t)strm * BSROWS) * Dm + h * DhD);
  };

  // Q fragments: qx from rows [0,4096), qy from rows [4096,8192).
  const size_t rowQx = (size_t)(b * Sq + qBase + lm);
  const unsigned* qxp = (const unsigned*)(Q + rowQx * Dm + h * DhD);
  const unsigned* qyp = (const unsigned*)(Q + (rowQx + BSROWS) * Dm + h * DhD);
  AFrag qxA[2], qyA[2];
  #pragma unroll
  for (int ds_ = 0; ds_ < 2; ++ds_) {
    #pragma unroll
    for (int v = 0; v < 8; ++v) {
      const int kb = ((v >> 2) << 4) + ((v & 3) << 1) + (half << 3);
      const int d = ds_ * 32 + kb;
      qxA[ds_].u[v] = qxp[d >> 1];
      qyA[ds_].u[v] = qyp[d >> 1];
    }
  }

  float mrun[8], srun[8];
  #pragma unroll
  for (int v = 0; v < 8; ++v) { mrun[v] = -1e30f; srun[v] = 0.0f; }
  v8f ax[4] = {}, ay[4] = {};

  const int keyL = tid >> 2, db = (tid & 3) << 4;   // loader split: 32x4

  // Prologue: kick off TDM for chunk 0.
  if (wave == 0) {
    tdm_load_2d_bf16(ldsVx[0], gaV(0, 0), DhD, 32, Dm, MQ, Dm);
    tdm_load_2d_bf16(ldsVy[0], gaV(0, 1), DhD, 32, Dm, MQ, Dm);
  }

  for (int kc = 0; kc < Sq; kc += 32) {
    const int cb = (kc >> 5) & 1;

    // ---- stage K chunks (transposed) into LDS, all threads ----
    const size_t rKx = (size_t)(b * Sq + kc + keyL);
    const size_t rKy = rKx + BSROWS;
    const uint4* kxp = (const uint4*)(K + rKx * Dm + h * DhD + db);
    const uint4* kyp = (const uint4*)(K + rKy * Dm + h * DhD + db);
    uint4 wx[2] = { kxp[0], kxp[1] };
    uint4 wy[2] = { kyp[0], kyp[1] };
    const unsigned* ux = (const unsigned*)wx;
    const unsigned* uy = (const unsigned*)wy;
    #pragma unroll
    for (int i = 0; i < 8; ++i) {
      Kxt[db + 2 * i][keyL]     = (unsigned short)(ux[i] & 0xFFFFu);
      Kxt[db + 2 * i + 1][keyL] = (unsigned short)(ux[i] >> 16);
      Kyt[db + 2 * i][keyL]     = (unsigned short)(uy[i] & 0xFFFFu);
      Kyt[db + 2 * i + 1][keyL] = (unsigned short)(uy[i] >> 16);
    }

    // ---- pipeline: issue TDM for next chunk, then wait for current ----
    if (wave == 0) {
      if (kc + 32 < Sq) {
        tdm_load_2d_bf16(ldsVx[cb ^ 1], gaV(kc + 32, 0), DhD, 32, Dm, MQ, Dm);
        tdm_load_2d_bf16(ldsVy[cb ^ 1], gaV(kc + 32, 1), DhD, 32, Dm, MQ, Dm);
        __builtin_amdgcn_s_wait_tensorcnt(2);   // current pair complete
      } else {
        __builtin_amdgcn_s_wait_tensorcnt(0);
      }
    }
    __syncthreads();

    // ---- scores: s = (qx.kx + qy.ky) * 0.5 / sqrt(Dh) ----
    v8f sc[2];
    #pragma unroll
    for (int kt = 0; kt < 2; ++kt) {
      v8f s = {};
      #pragma unroll
      for (int ds_ = 0; ds_ < 2; ++ds_) {
        AFrag bkx, bky;
        #pragma unroll
        for (int v = 0; v < 8; ++v) {
          const int kb = ds_ * 32 + ((v >> 2) << 4) + ((v & 3) << 1) + (half << 3);
          bkx.u[v] = (unsigned)Kxt[kb][(kt << 4) + lm] | ((unsigned)Kxt[kb + 1][(kt << 4) + lm] << 16);
          bky.u[v] = (unsigned)Kyt[kb][(kt << 4) + lm] | ((unsigned)Kyt[kb + 1][(kt << 4) + lm] << 16);
        }
        s = wmma_bf16(qxA[ds_].v, bkx.v, s);
        s = wmma_bf16(qyA[ds_].v, bky.v, s);
      }
      sc[kt] = s * 0.0625f;   // 0.5 / sqrt(64)
    }

    // ---- online softmax update (row stats replicated per 16-lane half) ----
    #pragma unroll
    for (int v = 0; v < 8; ++v) {
      float t = fmaxf(sc[0][v], sc[1][v]);
      t = fmaxf(t, __shfl_xor(t, 1, 32));
      t = fmaxf(t, __shfl_xor(t, 2, 32));
      t = fmaxf(t, __shfl_xor(t, 4, 32));
      t = fmaxf(t, __shfl_xor(t, 8, 32));
      const float mn = fmaxf(mrun[v], t);
      const float sf = __expf(mrun[v] - mn);
      mrun[v] = mn;
      srun[v] *= sf;
      #pragma unroll
      for (int dt = 0; dt < 4; ++dt) { ax[dt][v] *= sf; ay[dt][v] *= sf; }
      const float p0 = __expf(sc[0][v] - mn);
      const float p1 = __expf(sc[1][v] - mn);
      float ps = p0 + p1;
      ps += __shfl_xor(ps, 1, 32);
      ps += __shfl_xor(ps, 2, 32);
      ps += __shfl_xor(ps, 4, 32);
      ps += __shfl_xor(ps, 8, 32);
      srun[v] += ps;
      Ps[wave][v + (half << 3)][lm]      = f2bf(p0);
      Ps[wave][v + (half << 3)][16 + lm] = f2bf(p1);
    }
    __builtin_amdgcn_sched_barrier(0);
    __builtin_amdgcn_wave_barrier();  // wave-private LDS RAW; DS ops in-order

    // ---- P (A-layout) @ V ----
    AFrag pA;
    #pragma unroll
    for (int v = 0; v < 8; ++v) {
      const int kb = ((v >> 2) << 4) + ((v & 3) << 1) + (half << 3);
      pA.u[v] = *(const unsigned*)&Ps[wave][lm][kb];
    }
    #pragma unroll
    for (int dt = 0; dt < 4; ++dt) {
      AFrag bvx, bvy;
      #pragma unroll
      for (int v = 0; v < 8; ++v) {
        const int kb = ((v >> 2) << 4) + ((v & 3) << 1) + (half << 3);
        bvx.u[v] = (unsigned)Vxs[cb][kb][(dt << 4) + lm] | ((unsigned)Vxs[cb][kb + 1][(dt << 4) + lm] << 16);
        bvy.u[v] = (unsigned)Vys[cb][kb][(dt << 4) + lm] | ((unsigned)Vys[cb][kb + 1][(dt << 4) + lm] << 16);
      }
      ax[dt] = wmma_bf16(pA.v, bvx.v, ax[dt]);
      ay[dt] = wmma_bf16(pA.v, bvy.v, ay[dt]);
    }
    __syncthreads();
  }

  // ---- normalize + store bf16 ----
  #pragma unroll
  for (int v = 0; v < 8; ++v) {
    const float inv = 1.0f / srun[v];
    const size_t r = (size_t)(b * Sq + qBase + v + (half << 3));
    #pragma unroll
    for (int dt = 0; dt < 4; ++dt) {
      const int c = h * DhD + (dt << 4) + lm;
      A1[r * Dm + c] = f2bf(ax[dt][v] * inv);
      A2[r * Dm + c] = f2bf(ay[dt][v] * inv);
    }
  }
}

// ---------------------------------------------------------------------------
// Kernel 3: output projection. out[M=4096,1024] = A(bf16) @ W + bias + res.
// A tiles via pipelined TDM into ping-pong LDS; W staged manually (f32->bf16).
// ---------------------------------------------------------------------------
__global__ void __launch_bounds__(128)
gemm_out(const unsigned short* __restrict__ A, const float* __restrict__ W,
         const float* __restrict__ bias, const float* __restrict__ res,
         float* __restrict__ out)
{
  __shared__ __align__(32) unsigned short As[2][64][32];
  __shared__ __align__(32) unsigned short Bs[2][32][64];
  const int tid = threadIdx.x, lane = tid & 31, wave = tid >> 5;
  const int half = lane >> 4, lm = lane & 15;
  const int mBase = blockIdx.x * 64, nBase = blockIdx.y * 64;

  v8f acc[4] = {};
  const int brow = tid >> 2, bcb = (tid & 3) << 4;

  const unsigned ldsA[2] = { (unsigned)(uintptr_t)(void*)&As[0][0][0],
                             (unsigned)(uintptr_t)(void*)&As[1][0][0] };
  auto tdmA = [&](int buf, int kk) {
    const unsigned long long ga =
        (unsigned long long)(uintptr_t)(A + (size_t)mBase * Dm + kk);
    tdm_load_2d_bf16(ldsA[buf], ga, 32, 64, Dm, BSROWS, Dm);
  };
  auto loadB = [&](int buf, int kk) {
    const float* bp = W + (size_t)(kk + brow) * Dm + nBase + bcb;
    __builtin_prefetch(bp + 32 * Dm, 0, 3);
    #pragma unroll
    for (int i = 0; i < 8; ++i)
      *(unsigned*)&Bs[buf][brow][bcb + 2 * i] = f2bf_pk(bp[2 * i], bp[2 * i + 1]);
  };

  if (wave == 0) tdmA(0, 0);
  loadB(0, 0);
  if (wave == 0) __builtin_amdgcn_s_wait_tensorcnt(0);
  __syncthreads();

  for (int kk = 0; kk < Dm; kk += 32) {
    const int cur = (kk >> 5) & 1;
    if (kk + 32 < Dm) {
      if (wave == 0) tdmA(cur ^ 1, kk + 32);
      loadB(cur ^ 1, kk + 32);
    }

    AFrag a;
    const int m = wave * 16 + lm;
    #pragma unroll
    for (int v = 0; v < 8; ++v) {
      const int kb = ((v >> 2) << 4) + ((v & 3) << 1) + (half << 3);
      a.u[v] = *(const unsigned*)&As[cur][m][kb];
    }
    #pragma unroll
    for (int nt = 0; nt < 4; ++nt) {
      AFrag bb;
      #pragma unroll
      for (int v = 0; v < 8; ++v) {
        const int kb = ((v >> 2) << 4) + ((v & 3) << 1) + (half << 3);
        bb.u[v] = (unsigned)Bs[cur][kb][(nt << 4) + lm] | ((unsigned)Bs[cur][kb + 1][(nt << 4) + lm] << 16);
      }
      acc[nt] = wmma_bf16(a.v, bb.v, acc[nt]);
    }
    if (wave == 0) __builtin_amdgcn_s_wait_tensorcnt(0);
    __syncthreads();
  }

  #pragma unroll
  for (int nt = 0; nt < 4; ++nt) {
    #pragma unroll
    for (int v = 0; v < 8; ++v) {
      const size_t r = (size_t)(mBase + wave * 16 + v + (half << 3));
      const int c = nBase + (nt << 4) + lm;
      const float rv = __builtin_nontemporal_load(&res[r * Dm + c]);
      __builtin_nontemporal_store(acc[nt][v] + bias[c] + rv, &out[r * Dm + c]);
    }
  }
}

// ---------------------------------------------------------------------------
extern "C" void kernel_launch(void* const* d_in, const int* in_sizes, int n_in,
                              void* d_out, int out_size, void* d_ws, size_t ws_size,
                              hipStream_t stream) {
  (void)in_sizes; (void)n_in; (void)out_size; (void)ws_size;
  const float* x   = (const float*)d_in[0];
  const float* y   = (const float*)d_in[1];
  const float* Wq  = (const float*)d_in[2];
  const float* Wk  = (const float*)d_in[3];
  const float* Wv  = (const float*)d_in[4];
  const float* Wox = (const float*)d_in[5];
  const float* box = (const float*)d_in[6];
  const float* Woy = (const float*)d_in[7];
  const float* boy = (const float*)d_in[8];

  unsigned short* Qb = (unsigned short*)d_ws;           // [8192,1024] bf16
  unsigned short* Kb = Qb + (size_t)MQ * Dm;            // [8192,1024] bf16
  unsigned short* Vb = Kb + (size_t)MQ * Dm;            // [8192,1024] bf16
  unsigned short* A1 = Vb + (size_t)MQ * Dm;            // [4096,1024] bf16
  unsigned short* A2 = A1 + (size_t)BSROWS * Dm;        // [4096,1024] bf16
  float* out1 = (float*)d_out;
  float* out2 = out1 + (size_t)BSROWS * Dm;

  dim3 gProj(MQ / 64, Dm / 64);       // (128, 16)
  gemm_qkv<<<gProj, 128, 0, stream>>>(y, x, Wq, Qb);  // cross-wired Q
  gemm_qkv<<<gProj, 128, 0, stream>>>(x, y, Wk, Kb);
  gemm_qkv<<<gProj, 128, 0, stream>>>(x, y, Wv, Vb);

  attn<<<dim3(Sq / 64, Bz * Hh), 128, 0, stream>>>(Qb, Kb, Vb, A1, A2);

  dim3 gOut(BSROWS / 64, Dm / 64);    // (64, 16)
  gemm_out<<<gOut, 128, 0, stream>>>(A1, Wox, box, x, out1);
  gemm_out<<<gOut, 128, 0, stream>>>(A2, Woy, boy, y, out2);
}
